// BatchTGAT_86474871538516
// MI455X (gfx1250) — matrix operand order repaired and checked by reading, not compiled, not run on hardware
//
#include <hip/hip_runtime.h>
#include <hip/hip_bf16.h>

// ---------------------------------------------------------------------------
// MI455X (gfx1250) TGAT implementation.
// bf16 WMMA (v_wmma_f32_16x16x32_bf16) for all M>=16 GEMMs, TDM tensor_load
// for the contiguous per-node rel tiles, fp32 VALU for M=1 vector ops.
// ---------------------------------------------------------------------------

typedef __bf16          v16bf __attribute__((ext_vector_type(16)));
typedef float           v8f   __attribute__((ext_vector_type(8)));
typedef unsigned short  v8us  __attribute__((ext_vector_type(8)));
typedef unsigned int    u32x4 __attribute__((ext_vector_type(4)));
typedef int             i32x4 __attribute__((ext_vector_type(4)));
typedef int             i32x8 __attribute__((ext_vector_type(8)));

#define KNBR 64
#define DMODEL 192
#define HDIM 128
#define NNODES 100000

__device__ __forceinline__ unsigned short f2bf(float f) {
  unsigned int u = __builtin_bit_cast(unsigned int, f);
  u += 0x7FFFu + ((u >> 16) & 1u);          // round-to-nearest-even
  return (unsigned short)(u >> 16);
}
__device__ __forceinline__ float bf2f(unsigned short h) {
  unsigned int u = ((unsigned int)h) << 16;
  return __builtin_bit_cast(float, u);
}

union ABpack { v16bf v; v8us h[2]; };

// A tile (16x32 bf16) from LDS, per documented 16-bit A layout:
// lanes 0-15: row = lane,   K chunks {kb..kb+7, kb+16..kb+23}
// lanes16-31: row = lane-16,K chunks {kb+8..kb+15, kb+24..kb+31}
__device__ __forceinline__ v16bf load_A_lds(const unsigned short* base, int lane,
                                            int m0, int kbase, int stride) {
  int r  = m0 + (lane & 15);
  int hi = lane >> 4;
  const unsigned short* p = base + r * stride + kbase + hi * 8;
  ABpack u;
  u.h[0] = *(const v8us*)(p);
  u.h[1] = *(const v8us*)(p + 16);
  return u.v;
}

// B tile (32x16 bf16) from a transposed bf16 weight WT[fo][fi]:
// lanes 0-15: N = n0+lane, K chunks {kb..kb+7, kb+16..kb+23}
// lanes16-31: N = n0+lane-16, K chunks {kb+8..kb+15, kb+24..kb+31}
__device__ __forceinline__ v16bf load_B_glb(const unsigned short* WT, int lane,
                                            int n0, int kbase, int fi) {
  int nn = n0 + (lane & 15);
  int k0 = kbase + (lane >> 4) * 8;
  const unsigned short* p = WT + nn * fi + k0;
  ABpack u;
  u.h[0] = *(const v8us*)(p);
  u.h[1] = *(const v8us*)(p + 16);
  return u.v;
}

__device__ __forceinline__ v8f wmma_bf16(v16bf a, v16bf b, v8f c) {
  return __builtin_amdgcn_wmma_f32_16x16x32_bf16(false, a, false, b,
                                                 (short)0, c, false, false);
}

// ---------------------------------------------------------------------------
// TDM: 2-D fp32 tile load (dim0 x dim1 elements, row stride stride0 elements)
// from global to LDS. Descriptor per CDNA5 ISA §8 (group0/group1).
// ---------------------------------------------------------------------------
__device__ __forceinline__ void tdm_load_2d_f32(unsigned lds_off, const float* gptr,
                                                unsigned dim0, unsigned dim1,
                                                unsigned stride0) {
#if __has_builtin(__builtin_amdgcn_tensor_load_to_lds)
  unsigned long long ga = (unsigned long long)gptr;
  u32x4 g0;
  g0.x = 1u;                                             // count=1, user mode
  g0.y = lds_off;                                        // lds_addr
  g0.z = (unsigned)(ga & 0xFFFFFFFFu);                   // global_addr[31:0]
  g0.w = ((unsigned)(ga >> 32) & 0x01FFFFFFu) | (2u << 30); // addr[56:32] | type=2
  i32x8 g1;
  g1[0] = (int)(2u << 16);                               // data_size = 4B
  g1[1] = (int)((dim0 & 0xFFFFu) << 16);                 // tensor_dim0[15:0]
  g1[2] = (int)(((dim0 >> 16) & 0xFFFFu) | ((dim1 & 0xFFFFu) << 16)); // dim0 hi | dim1 lo
  g1[3] = (int)(((dim1 >> 16) & 0xFFFFu) | ((dim0 & 0xFFFFu) << 16)); // dim1 hi | tile_dim0
  g1[4] = (int)(dim1 & 0xFFFFu);                         // tile_dim1 (tile_dim2=0)
  g1[5] = (int)stride0;                                  // tensor_dim0_stride[31:0]
  g1[6] = 0;
  g1[7] = 0;
  i32x4 z4 = {0, 0, 0, 0};
#if defined(__clang_major__) && __clang_major__ >= 23
  i32x8 z8 = {0, 0, 0, 0, 0, 0, 0, 0};
  __builtin_amdgcn_tensor_load_to_lds(g0, g1, z4, z4, z8, 0);
#else
  __builtin_amdgcn_tensor_load_to_lds(g0, g1, z4, z4, 0);
#endif
#else
  (void)lds_off; (void)gptr; (void)dim0; (void)dim1; (void)stride0;
#endif
}

// ---------------------------------------------------------------------------
// Weight convert + transpose: W[fi][fo] fp32 -> WT[fo][fi] bf16
// ---------------------------------------------------------------------------
__global__ void wconv_kernel(const float* __restrict__ W,
                             unsigned short* __restrict__ WT, int fi, int fo) {
  int e = blockIdx.x * blockDim.x + threadIdx.x;
  if (e >= fi * fo) return;
  int n = e / fi, k = e - n * fi;
  WT[e] = f2bf(W[k * fo + n]);
}

// ---------------------------------------------------------------------------
// hx = relu(x @ Wp + bp) for all NNODES rows -> bf16 [NNODES,128]
// One WG = 64 rows, 4 waves x 2 N-tiles each, WMMA bf16.
// ---------------------------------------------------------------------------
__global__ __launch_bounds__(128) void proj_all_kernel(
    const float* __restrict__ x, const unsigned short* __restrict__ WpT,
    const float* __restrict__ bp, unsigned short* __restrict__ hx, int ntot) {
  __shared__ unsigned short xs[64][136];   // bf16, padded stride (272B, 16B mult)
  const int tid = threadIdx.x, lane = tid & 31, w = tid >> 5;
  const int m0 = blockIdx.x * 64;

  for (int i = tid; i < 64 * 32; i += 128) {           // 32 float4 per row
    int r = i >> 5, c4 = i & 31;
    int row = m0 + r; if (row >= ntot) row = ntot - 1;
    float4 f = ((const float4*)x)[(size_t)row * 32 + c4];
    xs[r][c4 * 4 + 0] = f2bf(f.x);
    xs[r][c4 * 4 + 1] = f2bf(f.y);
    xs[r][c4 * 4 + 2] = f2bf(f.z);
    xs[r][c4 * 4 + 3] = f2bf(f.w);
  }
  __syncthreads();

  v8f acc[2][4];
#pragma unroll
  for (int nt = 0; nt < 2; ++nt)
#pragma unroll
    for (int m = 0; m < 4; ++m)
#pragma unroll
      for (int j = 0; j < 8; ++j) acc[nt][m][j] = 0.f;

#pragma unroll 1
  for (int kk = 0; kk < 4; ++kk) {
    v16bf b0 = load_B_glb(WpT, lane, (2 * w + 0) * 16, kk * 32, 128);
    v16bf b1 = load_B_glb(WpT, lane, (2 * w + 1) * 16, kk * 32, 128);
#pragma unroll
    for (int m = 0; m < 4; ++m) {
      v16bf a = load_A_lds(&xs[0][0], lane, m * 16, kk * 32, 136);
      acc[0][m] = wmma_bf16(a, b0, acc[0][m]);
      acc[1][m] = wmma_bf16(a, b1, acc[1][m]);
    }
  }
  const int hi = lane >> 4;
#pragma unroll
  for (int nt = 0; nt < 2; ++nt) {
    int col = (2 * w + nt) * 16 + (lane & 15);
    float b = bp[col];
#pragma unroll
    for (int m = 0; m < 4; ++m)
#pragma unroll
      for (int j = 0; j < 8; ++j) {
        int row = m0 + m * 16 + hi * 8 + j;
        if (row < ntot) {
          float v = acc[nt][m][j] + b;
          hx[(size_t)row * 128 + col] = f2bf(v > 0.f ? v : 0.f);
        }
      }
  }
}

// ---------------------------------------------------------------------------
// Fused TGAT layer. One node per 128-thread WG (4 waves).
//  src row  n : src_idx ? src_tab[src_idx[n]]   : src_tab[n]        (bf16[128])
//  neigh r,n  : neigh_idx ? neigh_tab[neigh_idx[n*64+r]] : neigh_tab[n*64+r]
// ---------------------------------------------------------------------------
__global__ __launch_bounds__(128) void tgat_layer_kernel(
    const unsigned short* __restrict__ src_tab, const int* __restrict__ src_idx,
    const unsigned short* __restrict__ neigh_tab, const int* __restrict__ neigh_idx,
    const float* __restrict__ t, const float* __restrict__ rel,
    const float* __restrict__ tk_w, const float* __restrict__ tk_b,
    const float* __restrict__ src_param,
    const unsigned short* __restrict__ WqT, const float* __restrict__ bq,
    const unsigned short* __restrict__ WkT, const float* __restrict__ bk,
    const unsigned short* __restrict__ WvT, const float* __restrict__ bv,
    const unsigned short* __restrict__ WoT, const float* __restrict__ bo,
    const unsigned short* __restrict__ Wm1T, const float* __restrict__ bm1,
    const unsigned short* __restrict__ Wm2T, const float* __restrict__ bm2,
    unsigned short* __restrict__ out) {
  __shared__ unsigned short kv[64][200];   // [K,192] bf16, padded
  __shared__ unsigned short km[64][200];   // K-projection output
  __shared__ unsigned short vm[64][200];   // V-projection output
  __shared__ float rel_s[64 * 32];         // TDM destination (fp32)
  __shared__ float srcin[192];
  __shared__ float qv[192];
  __shared__ float tsrow[64];
  __shared__ float sc[256];                // [4 heads][64]
  __shared__ float red[8];
  __shared__ float valv[192];
  __shared__ float aout[192];
  __shared__ float mh[128];
  __shared__ float tkw_s[32], tkb_s[32];

  const int tid = threadIdx.x, lane = tid & 31, w = tid >> 5;
  const int n = blockIdx.x;

  // --- phase 0: kick off TDM for rel tile, stage small constants ------------
  if (tid == 0) {
    tdm_load_2d_f32((unsigned)(size_t)rel_s, rel + (size_t)n * (KNBR * 32),
                    32u, 64u, 32u);
    __builtin_prefetch(WkT, 0, 3);
    __builtin_prefetch(WvT, 0, 3);
  }
  if (tid < 32) { tkw_s[tid] = tk_w[tid]; tkb_s[tid] = tk_b[tid]; }
  if (tid < 64) tsrow[tid] = t[(size_t)n * KNBR + tid];
  __syncthreads();

  // --- phase 1: t.max reduction + source input vector ----------------------
  if (tid == 0) {
    float mx = tsrow[0];
    for (int r = 1; r < 64; ++r) mx = fmaxf(mx, tsrow[r]);
    red[0] = mx;
  }
  {
    int srow = src_idx ? src_idx[n] : n;
    srcin[tid] = bf2f(src_tab[(size_t)srow * 128 + tid]);   // src_x
    if (tid < 32)       srcin[128 + tid] = cosf(tkb_s[tid]); // time-enc of 0
    else if (tid < 64)  srcin[128 + tid] = src_param[tid - 32];
  }
  __syncthreads();

  // --- phase 2: build kv = [neigh_h | time_enc | rel] in bf16 --------------
  for (int k = 0; k < 64; ++k) {                         // gathered bf16 rows
    int row = neigh_idx ? neigh_idx[(size_t)n * KNBR + k] : (n * KNBR + k);
    kv[k][tid] = neigh_tab[(size_t)row * 128 + tid];
  }
  {
    float tmax = red[0];
    for (int i = tid; i < 64 * 32; i += 128) {           // harmonic time enc
      int r = i >> 5, c = i & 31;
      float ts = tmax - tsrow[r];
      kv[r][128 + c] = f2bf(cosf(ts * tkw_s[c] + tkb_s[c]));
    }
  }
  if (tid < 32) __builtin_amdgcn_s_wait_tensorcnt(0);    // wave0 fences TDM
  __syncthreads();
  for (int i = tid; i < 64 * 32; i += 128) {             // rel -> bf16
    int r = i >> 5, c = i & 31;
    kv[r][160 + c] = f2bf(rel_s[i]);
  }
  __syncthreads();

  // --- phase 3: K and V projections via WMMA bf16 --------------------------
  // wave w owns N-tiles {3w,3w+1,3w+2}; 4 M-tiles; 6 K-slices of 32.
  for (int pv = 0; pv < 2; ++pv) {
    const unsigned short* WT = pv ? WvT : WkT;
    const float* bias = pv ? bv : bk;
    unsigned short(*dst)[200] = pv ? vm : km;
    v8f acc[3][4];
#pragma unroll
    for (int nt = 0; nt < 3; ++nt)
#pragma unroll
      for (int m = 0; m < 4; ++m)
#pragma unroll
        for (int j = 0; j < 8; ++j) acc[nt][m][j] = 0.f;
#pragma unroll 1
    for (int kk = 0; kk < 6; ++kk) {
      v16bf b0 = load_B_glb(WT, lane, (w * 3 + 0) * 16, kk * 32, 192);
      v16bf b1 = load_B_glb(WT, lane, (w * 3 + 1) * 16, kk * 32, 192);
      v16bf b2 = load_B_glb(WT, lane, (w * 3 + 2) * 16, kk * 32, 192);
#pragma unroll
      for (int m = 0; m < 4; ++m) {
        v16bf a = load_A_lds(&kv[0][0], lane, m * 16, kk * 32, 200);
        acc[0][m] = wmma_bf16(a, b0, acc[0][m]);
        acc[1][m] = wmma_bf16(a, b1, acc[1][m]);
        acc[2][m] = wmma_bf16(a, b2, acc[2][m]);
      }
    }
    const int hi = lane >> 4;
#pragma unroll
    for (int nt = 0; nt < 3; ++nt) {
      int col = (w * 3 + nt) * 16 + (lane & 15);
      float bcol = bias[col];
#pragma unroll
      for (int m = 0; m < 4; ++m)
#pragma unroll
        for (int j = 0; j < 8; ++j)
          dst[m * 16 + hi * 8 + j][col] = f2bf(acc[nt][m][j] + bcol);
    }
  }
  __syncthreads();

  // --- phase 4: q projection (M=1, VALU) -----------------------------------
  for (int nn = tid; nn < 192; nn += 128) {
    float s = bq[nn];
    const unsigned short* wrow = WqT + nn * 192;
    for (int k2 = 0; k2 < 192; ++k2) s += srcin[k2] * bf2f(wrow[k2]);
    qv[nn] = s;
  }
  __syncthreads();

  // --- phase 5: scores = q.k / sqrt(48), softmax over K=64 per head --------
  for (int e = tid; e < 256; e += 128) {
    int h = e >> 6, r = e & 63;
    float s = 0.f;
    for (int d = 0; d < 48; ++d) s += qv[h * 48 + d] * bf2f(km[r][h * 48 + d]);
    sc[e] = s * 0.14433756729740643f;   // 1/sqrt(48)
  }
  __syncthreads();
  if (tid < 4) {
    float mx = sc[tid * 64];
    for (int r = 1; r < 64; ++r) mx = fmaxf(mx, sc[tid * 64 + r]);
    float sum = 0.f;
    for (int r = 0; r < 64; ++r) sum += expf(sc[tid * 64 + r] - mx);
    red[tid] = mx; red[4 + tid] = 1.f / sum;
  }
  __syncthreads();
  for (int e = tid; e < 256; e += 128) {
    int h = e >> 6;
    sc[e] = expf(sc[e] - red[h]) * red[4 + h];
  }
  __syncthreads();

  // --- phase 6: val = attn @ V ---------------------------------------------
  for (int nn = tid; nn < 192; nn += 128) {
    int h = nn / 48;
    float s = 0.f;
    for (int r = 0; r < 64; ++r) s += sc[h * 64 + r] * bf2f(vm[r][nn]);
    valv[nn] = s;
  }
  __syncthreads();

  // --- phase 7: output proj val @ Wo + bo ----------------------------------
  for (int nn = tid; nn < 192; nn += 128) {
    float s = bo[nn];
    const unsigned short* wrow = WoT + nn * 192;
    for (int k2 = 0; k2 < 192; ++k2) s += valv[k2] * bf2f(wrow[k2]);
    aout[nn] = s;
  }
  __syncthreads();

  // --- phase 8: MLP: relu([src_x | val] @ Wm1 + bm1) @ Wm2 + bm2 -----------
  {
    float s = bm1[tid];
    const unsigned short* wrow = Wm1T + tid * 320;
    for (int k2 = 0; k2 < 128; ++k2) s += srcin[k2] * bf2f(wrow[k2]);
    for (int k2 = 0; k2 < 192; ++k2) s += aout[k2] * bf2f(wrow[128 + k2]);
    mh[tid] = s > 0.f ? s : 0.f;
  }
  __syncthreads();
  {
    float s = bm2[tid];
    const unsigned short* wrow = Wm2T + tid * 128;
    for (int k2 = 0; k2 < 128; ++k2) s += mh[k2] * bf2f(wrow[k2]);
    out[(size_t)n * 128 + tid] = f2bf(s);
  }
}

// ---------------------------------------------------------------------------
// Final: out[b,o] = h2[b] @ Wout + bout   (256 x 64, fp32)
// ---------------------------------------------------------------------------
__global__ void out_kernel(const unsigned short* __restrict__ h2,
                           const float* __restrict__ Wout,
                           const float* __restrict__ bout,
                           float* __restrict__ out) {
  int g = blockIdx.x * blockDim.x + threadIdx.x;   // 16384
  int b = g >> 6, o = g & 63;
  float s = bout[o];
  for (int k = 0; k < 128; ++k) s += bf2f(h2[b * 128 + k]) * Wout[k * 64 + o];
  out[g] = s;
}

// ---------------------------------------------------------------------------
extern "C" void kernel_launch(void* const* d_in, const int* in_sizes, int n_in,
                              void* d_out, int out_size, void* d_ws, size_t ws_size,
                              hipStream_t stream) {
  (void)in_sizes; (void)n_in; (void)out_size; (void)ws_size;
  const float* x         = (const float*)d_in[0];
  const int*   batch     = (const int*)d_in[1];
  const int*   nbr1_idx  = (const int*)d_in[2];
  const float* nbr1_t    = (const float*)d_in[3];
  const float* nbr1_rel  = (const float*)d_in[4];
  const int*   nbr2_idx  = (const int*)d_in[5];
  const float* nbr2_t    = (const float*)d_in[6];
  const float* nbr2_rel  = (const float*)d_in[7];
  const int*   nbr1n_idx = (const int*)d_in[8];
  const float* nbr1n_t   = (const float*)d_in[9];
  const float* nbr1n_rel = (const float*)d_in[10];
  const float* Wp  = (const float*)d_in[11]; const float* bp  = (const float*)d_in[12];
  const float* tk_w = (const float*)d_in[13]; const float* tk_b = (const float*)d_in[14];
  const float* Wq  = (const float*)d_in[15]; const float* bq  = (const float*)d_in[16];
  const float* Wk  = (const float*)d_in[17]; const float* bk  = (const float*)d_in[18];
  const float* Wv  = (const float*)d_in[19]; const float* bv  = (const float*)d_in[20];
  const float* Wo  = (const float*)d_in[21]; const float* bo  = (const float*)d_in[22];
  const float* Wm1 = (const float*)d_in[23]; const float* bm1 = (const float*)d_in[24];
  const float* Wm2 = (const float*)d_in[25]; const float* bm2 = (const float*)d_in[26];
  const float* Wout = (const float*)d_in[27]; const float* bout = (const float*)d_in[28];
  const float* src_param = (const float*)d_in[29];

  char* ws = (char*)d_ws;
  size_t off = 0;
  auto take = [&](size_t bytes) {
    size_t cur = off;
    off += (bytes + 255) & ~(size_t)255;
    return cur;
  };
  unsigned short* WpT  = (unsigned short*)(ws + take((size_t)128 * 128 * 2));
  unsigned short* WqT  = (unsigned short*)(ws + take((size_t)192 * 192 * 2));
  unsigned short* WkT  = (unsigned short*)(ws + take((size_t)192 * 192 * 2));
  unsigned short* WvT  = (unsigned short*)(ws + take((size_t)192 * 192 * 2));
  unsigned short* WoT  = (unsigned short*)(ws + take((size_t)192 * 192 * 2));
  unsigned short* Wm1T = (unsigned short*)(ws + take((size_t)320 * 128 * 2));
  unsigned short* Wm2T = (unsigned short*)(ws + take((size_t)128 * 128 * 2));
  unsigned short* hx   = (unsigned short*)(ws + take((size_t)NNODES * 128 * 2));
  unsigned short* h1n  = (unsigned short*)(ws + take((size_t)16384 * 128 * 2));
  unsigned short* h1b  = (unsigned short*)(ws + take((size_t)256 * 128 * 2));
  unsigned short* h2   = (unsigned short*)(ws + take((size_t)256 * 128 * 2));

  // 1) weight convert + transpose to bf16 [fo][fi]
  wconv_kernel<<<(128 * 128 + 255) / 256, 256, 0, stream>>>(Wp, WpT, 128, 128);
  wconv_kernel<<<(192 * 192 + 255) / 256, 256, 0, stream>>>(Wq, WqT, 192, 192);
  wconv_kernel<<<(192 * 192 + 255) / 256, 256, 0, stream>>>(Wk, WkT, 192, 192);
  wconv_kernel<<<(192 * 192 + 255) / 256, 256, 0, stream>>>(Wv, WvT, 192, 192);
  wconv_kernel<<<(192 * 192 + 255) / 256, 256, 0, stream>>>(Wo, WoT, 192, 192);
  wconv_kernel<<<(320 * 128 + 255) / 256, 256, 0, stream>>>(Wm1, Wm1T, 320, 128);
  wconv_kernel<<<(128 * 128 + 255) / 256, 256, 0, stream>>>(Wm2, Wm2T, 128, 128);

  // 2) project ALL distinct nodes once: hx = relu(x@Wp+bp) (bf16)
  proj_all_kernel<<<(NNODES + 63) / 64, 128, 0, stream>>>(x, WpT, bp, hx, NNODES);

  // 3) layer-1 on the 16384 one-hop neighbors
  tgat_layer_kernel<<<16384, 128, 0, stream>>>(
      hx, nbr2_idx, hx, nbr1n_idx, nbr1n_t, nbr1n_rel, tk_w, tk_b, src_param,
      WqT, bq, WkT, bk, WvT, bv, WoT, bo, Wm1T, bm1, Wm2T, bm2, h1n);

  // 4) layer-1 on the 256 batch nodes
  tgat_layer_kernel<<<256, 128, 0, stream>>>(
      hx, batch, hx, nbr1_idx, nbr1_t, nbr1_rel, tk_w, tk_b, src_param,
      WqT, bq, WkT, bk, WvT, bv, WoT, bo, Wm1T, bm1, Wm2T, bm2, h1b);

  // 5) layer-2 (dense src/neigh: h1n is exactly [256,64,128] row-major)
  tgat_layer_kernel<<<256, 128, 0, stream>>>(
      h1b, nullptr, h1n, nullptr, nbr2_t, nbr2_rel, tk_w, tk_b, src_param,
      WqT, bq, WkT, bk, WvT, bv, WoT, bo, Wm1T, bm1, Wm2T, bm2, h2);

  // 6) output projection
  out_kernel<<<(256 * 64) / 256, 256, 0, stream>>>(h2, Wout, bout, (float*)d_out);
}
